// ConAttn_32942399160858
// MI455X (gfx1250) — matrix operand order repaired and checked by reading, not compiled, not run on hardware
//
#include <hip/hip_runtime.h>

#define C_   128
#define MID_ 32
#define N_   4096
#define B_   4
#define TSTR 136     // bf16 LDS row stride for 128 channels (pad 8 -> conflict-free-ish)
#define TROW 4100    // f32 LDS row stride for 4096 keys (pad 4 -> banks q*4+n)

typedef unsigned short u16;
typedef __attribute__((ext_vector_type(16))) __bf16 v16bf;
typedef __attribute__((ext_vector_type(8)))  float  v8f;

union FragBF { v16bf v; uint4 q[2]; u16 s[16]; };

static __device__ __forceinline__ u16 f2bf(float f) {
  union { __bf16 h; u16 s; } u; u.h = (__bf16)f; return u.s;
}
static __device__ __forceinline__ float bf2f(u16 b) {
  union { __bf16 h; u16 s; } u; u.s = b; return (float)u.h;
}
static __device__ __forceinline__ v8f zero8() {
  v8f z = {0.f,0.f,0.f,0.f,0.f,0.f,0.f,0.f}; return z;
}
static __device__ __forceinline__ v8f wmma_bf16(const FragBF& a, const FragBF& b, v8f c) {
  return __builtin_amdgcn_wmma_f32_16x16x32_bf16(false, a.v, false, b.v, (short)0, c, false, false);
}

// ---------------- kernel 0: weight conversion + ksum zero ----------------
__global__ __launch_bounds__(256) void prep_kernel(
    const float* __restrict__ qw, const float* __restrict__ vw,
    const float* __restrict__ linw,
    u16* __restrict__ qwb, u16* __restrict__ vwb, u16* __restrict__ w3b,
    float* __restrict__ ksum)
{
  int i = blockIdx.x * 256 + threadIdx.x;
  if (i < C_*C_) { qwb[i] = f2bf(qw[i]); vwb[i] = f2bf(vw[i]); }
  if (i < B_*C_) ksum[i] = 0.f;
  int nt = gridDim.x * 256;
  for (int j = i; j < 9*C_*C_; j += nt) {
    int tap = j / (C_*C_); int rem = j - tap*C_*C_;
    int co = rem >> 7; int ci = rem & 127;
    // lin_w [co][ci][ky][kx] -> w3b[tap][co][ci]  (tap = ky*3+kx)
    w3b[j] = f2bf(linw[(co*C_ + ci)*9 + tap]);
  }
}

// ---------------- kernel 1: q/v 1x1 convs (WMMA) + norms + k + heads ----------------
__global__ __launch_bounds__(256) void qv_kernel(
    const float* __restrict__ x,
    const float* __restrict__ qb,   const float* __restrict__ vb,
    const float* __restrict__ lw1w, const float* __restrict__ lw1b,
    const float* __restrict__ lw2w, const float* __restrict__ lw2b,
    const float* __restrict__ bw1w, const float* __restrict__ bw1b,
    const float* __restrict__ bw2w, const float* __restrict__ bw2b,
    const u16* __restrict__ qwb,  const u16* __restrict__ vwb,
    u16* __restrict__ qbf, u16* __restrict__ kbf, u16* __restrict__ vbf,
    float* __restrict__ ksum, float* __restrict__ wmap, float* __restrict__ bmap)
{
  __shared__ __align__(16) u16 xT[64*TSTR];   // [token][channel] bf16
  __shared__ __align__(16) u16 qT[64*TSTR];   // q result, token-major
  __shared__ float normv[64];
  __shared__ float rcpn[64];

  const int tid  = threadIdx.x;
  const int b    = blockIdx.x >> 6;
  const int tok0 = (blockIdx.x & 63) << 6;
  const float* xb = x + (size_t)b*C_*N_;

  for (int i = tid; i < 64*C_; i += 256) {
    int ch = i >> 6, t = i & 63;
    xT[t*TSTR + ch] = f2bf(xb[(size_t)ch*N_ + tok0 + t]);
  }
  if (tid < 64) normv[tid] = 0.f;
  __syncthreads();

  const int lane = tid & 31, wid = tid >> 5;
  const int lo = lane & 15, hi = lane >> 4;
  const int co_base = wid << 4;

  for (int sel = 0; sel < 2; ++sel) {
    const u16*   W    = sel ? vwb : qwb;
    const float* bias = sel ? vb  : qb;
    FragBF afr[4];
    const uint4* wrow = (const uint4*)(W + (co_base + lo)*C_);
#pragma unroll
    for (int kb = 0; kb < 4; ++kb) {          // A: V0-3 = ci0+hi*8.., V4-7 = +16
      afr[kb].q[0] = wrow[kb*4 + hi];
      afr[kb].q[1] = wrow[kb*4 + 2 + hi];
    }
#pragma unroll
    for (int tt = 0; tt < 4; ++tt) {
      const int tok = (tt << 4) + lo;
      const uint4* brow = (const uint4*)(xT + tok*TSTR);
      v8f acc = zero8();
#pragma unroll
      for (int kb = 0; kb < 4; ++kb) {        // B: V0-3 = ci0+hi*8.., V4-7 = +16
        FragBF bf_;
        bf_.q[0] = brow[kb*4 + hi];
        bf_.q[1] = brow[kb*4 + 2 + hi];
        acc = wmma_bf16(afr[kb], bf_, acc);
      }
      if (sel == 0) {
        float nrm = 0.f;
#pragma unroll
        for (int r = 0; r < 8; ++r) {
          int co = co_base + (hi<<3) + r;
          float v = acc[r] + bias[co];
          u16 h = f2bf(v);
          qT[tok*TSTR + co] = h;
          qbf[((size_t)(b*N_ + tok0 + tok))*C_ + co] = h;
          nrm += v*v;
        }
        atomicAdd(&normv[tok], nrm);
      } else {
#pragma unroll
        for (int r = 0; r < 8; ++r) {
          int co = co_base + (hi<<3) + r;
          float v = acc[r] + bias[co];
          vbf[((size_t)b*C_ + co)*N_ + tok0 + tok] = f2bf(v);
        }
      }
    }
  }
  __syncthreads();

  if (tid < 64) {
    float n = sqrtf(normv[tid]);
    rcpn[tid] = 1.f / fmaxf(n, 1e-4f);
  }
  __syncthreads();

  { // cosine-normalized keys + ksum (for linear mean-of-logits)
    int co = tid & 127, half = tid >> 7;
    float s = 0.f;
    for (int t = half*32; t < half*32 + 32; ++t) {
      float kv = bf2f(qT[t*TSTR + co]) * rcpn[t];
      u16 h = f2bf(kv);
      kbf[((size_t)(b*N_ + tok0 + t))*C_ + co] = h;
      s += bf2f(h);
    }
    atomicAdd(&ksum[b*C_ + co], s);
  }

  if (tid < 64) { // weight/bias heads, one token per thread (tiny)
    float wacc = lw2b[0], bacc = bw2b[0];
    for (int m = 0; m < MID_; ++m) {
      float a1 = lw1b[m], a2 = bw1b[m];
      for (int ci = 0; ci < C_; ++ci) {
        float qv = bf2f(qT[tid*TSTR + ci]);
        a1 = fmaf(lw1w[m*C_ + ci], qv, a1);
        a2 = fmaf(bw1w[m*C_ + ci], qv, a2);
      }
      a1 = a1 >= 0.f ? a1 : 0.2f*a1;
      a2 = a2 >= 0.f ? a2 : 0.2f*a2;
      wacc = fmaf(lw2w[m], a1, wacc);
      bacc = fmaf(bw2w[m], a2, bacc);
    }
    wmap[b*N_ + tok0 + tid] = wacc;
    bmap[b*N_ + tok0 + tid] = bacc;
  }
}

// ---------------- kernel 2: attention, 16 queries per block, full key strip in LDS ----------------
__global__ __launch_bounds__(256) void attn_kernel(
    const u16* __restrict__ qbf, const u16* __restrict__ kbf,
    const u16* __restrict__ vbf, const float* __restrict__ ksum,
    const float* __restrict__ wmap, const float* __restrict__ bmap,
    float* __restrict__ yv)
{
  extern __shared__ float smem[];
  float* tl    = smem;               // [16][TROW]  t = logits*sparse
  float* smean = smem + 16*TROW;
  float* sw    = smean + 16;
  float* sb    = sw + 16;
  float* smax  = sb + 16;
  float* srs   = smax + 16;
  float* red   = srs + 16;           // [16][17]

  const int tid  = threadIdx.x;
  const int b    = blockIdx.x >> 8;
  const int q0   = (blockIdx.x & 255) << 4;
  const int lane = tid & 31, wid = tid >> 5;
  const int lo = lane & 15, hi = lane >> 4;

  if (tid < 16) { // mean[m] = ksum . q[:,m] / N  (linearity of the mean)
    const u16* qrow = qbf + ((size_t)(b*N_ + q0 + tid))*C_;
    const float* ks = ksum + b*C_;
    float s = 0.f;
    for (int c = 0; c < C_; ++c) s = fmaf(ks[c], bf2f(qrow[c]), s);
    smean[tid] = s * (1.f/(float)N_);
    sw[tid] = wmap[b*N_ + q0 + tid];
    sb[tid] = bmap[b*N_ + q0 + tid];
  }
  __syncthreads();

  FragBF qf[4]; // query B-fragments, live in registers for all of phase 1
  {
    const uint4* qrow = (const uint4*)(qbf + ((size_t)(b*N_ + q0 + lo))*C_);
#pragma unroll
    for (int kb = 0; kb < 4; ++kb) {
      qf[kb].q[0] = qrow[kb*4 + hi];
      qf[kb].q[1] = qrow[kb*4 + 2 + hi];
    }
  }
  const float mq = smean[lo], wq = sw[lo], bq = sb[lo];

  // phase 1: t[n,q] = logits * relu(logits - mean*w + b) for all 4096 keys
  for (int kt = wid; kt < 256; kt += 8) {
    const uint4* krow = (const uint4*)(kbf + ((size_t)(b*N_ + (kt<<4) + lo))*C_);
    if (kt + 8 < 256)
      __builtin_prefetch((const void*)(kbf + ((size_t)(b*N_ + ((kt+8)<<4) + lo))*C_), 0, 1);
    v8f acc = zero8();
#pragma unroll
    for (int kb = 0; kb < 4; ++kb) {
      FragBF af;
      af.q[0] = krow[kb*4 + hi];
      af.q[1] = krow[kb*4 + 2 + hi];
      acc = wmma_bf16(af, qf[kb], acc);
    }
    float o[8];
#pragma unroll
    for (int r = 0; r < 8; ++r) {
      float lg = acc[r];
      float sp = fmaxf(fmaf(-mq, wq, lg) + bq, 0.f);
      o[r] = lg * sp;
    }
    float4* dst = (float4*)(tl + lo*TROW + (kt<<4) + (hi<<3));
    dst[0] = make_float4(o[0], o[1], o[2], o[3]);
    dst[1] = make_float4(o[4], o[5], o[6], o[7]);
  }
  __syncthreads();

  // phase 2: per-query max and sum-exp over all 4096 keys (scan LDS)
  {
    const int q = tid >> 4, part = tid & 15;
    const float4* row = (const float4*)(tl + q*TROW + part*256);
    float m = -3.0e38f;
    for (int i = 0; i < 64; ++i) {
      float4 v = row[i];
      m = fmaxf(m, fmaxf(fmaxf(v.x, v.y), fmaxf(v.z, v.w)));
    }
    red[q*17 + part] = m;
    __syncthreads();
    if (tid < 16) {
      float mm = red[tid*17];
      for (int i = 1; i < 16; ++i) mm = fmaxf(mm, red[tid*17 + i]);
      smax[tid] = mm;
    }
    __syncthreads();
    const float mx = smax[q];
    float s = 0.f;
    for (int i = 0; i < 64; ++i) {
      float4 v = row[i];
      s += __expf(v.x - mx) + __expf(v.y - mx) + __expf(v.z - mx) + __expf(v.w - mx);
    }
    red[q*17 + part] = s;
    __syncthreads();
    if (tid < 16) {
      float ss = 0.f;
      for (int i = 0; i < 16; ++i) ss += red[tid*17 + i];
      srs[tid] = 1.f / ss;
    }
    __syncthreads();
  }

  // phase 3: out[c,m] = sum_n v[c,n] * max(softmax(t)*mask, 1e-8); wave = c-tile
  {
    const float mx = smax[lo], rs = srs[lo];
    const u16* vrow = vbf + (size_t)b*C_*N_ + (size_t)((wid<<4) + lo)*N_;
    v8f acc = zero8();
    for (int k0 = 0; k0 < N_; k0 += 32) {
      const uint4* vr = (const uint4*)(vrow + k0);
      FragBF af;                        // A: v rows c, K = keys contiguous
      af.q[0] = vr[hi*2];
      af.q[1] = vr[hi*2 + 1];
      const float4* t4 = (const float4*)(tl + lo*TROW + k0 + (hi<<4));
      FragBF bf_;                       // B: attn' per (key,query)
#pragma unroll
      for (int j = 0; j < 4; ++j) {
        float4 tv = t4[j];
        float a0 = (tv.x != 0.f) ? fmaxf(__expf(tv.x - mx)*rs, 1e-8f) : 1e-8f;
        float a1 = (tv.y != 0.f) ? fmaxf(__expf(tv.y - mx)*rs, 1e-8f) : 1e-8f;
        float a2 = (tv.z != 0.f) ? fmaxf(__expf(tv.z - mx)*rs, 1e-8f) : 1e-8f;
        float a3 = (tv.w != 0.f) ? fmaxf(__expf(tv.w - mx)*rs, 1e-8f) : 1e-8f;
        bf_.s[j*4+0] = f2bf(a0); bf_.s[j*4+1] = f2bf(a1);
        bf_.s[j*4+2] = f2bf(a2); bf_.s[j*4+3] = f2bf(a3);
      }
      acc = wmma_bf16(af, bf_, acc);
    }
#pragma unroll
    for (int r = 0; r < 8; ++r) {
      int c = (wid<<4) + (hi<<3) + r;
      yv[((size_t)b*C_ + c)*N_ + q0 + lo] = acc[r];
    }
  }
}

// ---------------- kernel 3: 3x3 conv (9 WMMA taps) + bias + leaky + residual ----------------
__global__ __launch_bounds__(256) void conv_kernel(
    const float* __restrict__ yv, const float* __restrict__ x,
    const u16* __restrict__ w3b, const float* __restrict__ linb,
    float* __restrict__ out)
{
  __shared__ __align__(16) u16 yT[198*TSTR]; // 3 rows x 66 cols halo, token-major bf16
  const int tid = threadIdx.x;
  const int b = blockIdx.x >> 6;
  const int h = blockIdx.x & 63;

  for (int i = tid; i < 198*C_; i += 256) {
    int ch = i / 198, t = i - ch*198;
    int r = t / 66, c = t - r*66;
    int hh = h - 1 + r, cc = c - 1;
    float v = 0.f;
    if (hh >= 0 && hh < 64 && cc >= 0 && cc < 64)
      v = yv[((size_t)b*C_ + ch)*N_ + hh*64 + cc];
    yT[t*TSTR + ch] = f2bf(v);
  }
  __syncthreads();

  const int lane = tid & 31, wid = tid >> 5;
  const int lo = lane & 15, hi = lane >> 4;
  const int co_base = wid << 4;
#pragma unroll
  for (int tt = 0; tt < 4; ++tt) {
    v8f acc = zero8();
    for (int tap = 0; tap < 9; ++tap) {
      int ky = tap / 3, kx = tap - ky*3;
      const uint4* wr = (const uint4*)(w3b + ((size_t)tap*C_ + co_base + lo)*C_);
      int t_in = ky*66 + (tt<<4) + lo + kx;
      const uint4* br = (const uint4*)(yT + t_in*TSTR);
#pragma unroll
      for (int kb = 0; kb < 4; ++kb) {
        FragBF af, bf_;
        af.q[0]  = wr[kb*4 + hi];
        af.q[1]  = wr[kb*4 + 2 + hi];
        bf_.q[0] = br[kb*4 + hi];
        bf_.q[1] = br[kb*4 + 2 + hi];
        acc = wmma_bf16(af, bf_, acc);
      }
    }
    const int n = h*64 + (tt<<4) + lo;
#pragma unroll
    for (int r = 0; r < 8; ++r) {
      int co = co_base + (hi<<3) + r;
      float v = acc[r] + linb[co];
      v = v >= 0.f ? v : 0.2f*v;
      v += x[((size_t)b*C_ + co)*N_ + n];
      out[((size_t)b*C_ + co)*N_ + n] = v;
    }
  }
}

extern "C" void kernel_launch(void* const* d_in, const int* in_sizes, int n_in,
                              void* d_out, int out_size, void* d_ws, size_t ws_size,
                              hipStream_t stream) {
  const float* x    = (const float*)d_in[0];
  const float* qw   = (const float*)d_in[1];
  const float* qb   = (const float*)d_in[2];
  const float* vw   = (const float*)d_in[3];
  const float* vb   = (const float*)d_in[4];
  const float* lw1w = (const float*)d_in[5];
  const float* lw1b = (const float*)d_in[6];
  const float* lw2w = (const float*)d_in[7];
  const float* lw2b = (const float*)d_in[8];
  const float* bw1w = (const float*)d_in[9];
  const float* bw1b = (const float*)d_in[10];
  const float* bw2w = (const float*)d_in[11];
  const float* bw2b = (const float*)d_in[12];
  const float* linw = (const float*)d_in[13];
  const float* linb = (const float*)d_in[14];
  float* out = (float*)d_out;
  (void)in_sizes; (void)n_in; (void)out_size; (void)ws_size;

  char* w = (char*)d_ws;
  size_t off = 0;
  auto alloc = [&](size_t bytes) -> void* {
    void* p = w + off; off += (bytes + 255) & ~(size_t)255; return p;
  };
  u16*   qbf  = (u16*)  alloc((size_t)B_*N_*C_*2);
  u16*   kbf  = (u16*)  alloc((size_t)B_*N_*C_*2);
  u16*   vbf  = (u16*)  alloc((size_t)B_*N_*C_*2);
  float* yv   = (float*)alloc((size_t)B_*N_*C_*4);
  float* wmap = (float*)alloc((size_t)B_*N_*4);
  float* bmap = (float*)alloc((size_t)B_*N_*4);
  float* ksum = (float*)alloc((size_t)B_*C_*4);
  u16*   qwb  = (u16*)  alloc((size_t)C_*C_*2);
  u16*   vwb  = (u16*)  alloc((size_t)C_*C_*2);
  u16*   w3b  = (u16*)  alloc((size_t)9*C_*C_*2);

  prep_kernel<<<64, 256, 0, stream>>>(qw, vw, linw, qwb, vwb, w3b, ksum);
  qv_kernel<<<256, 256, 0, stream>>>(x, qb, vb, lw1w, lw1b, lw2w, lw2b,
                                     bw1w, bw1b, bw2w, bw2b, qwb, vwb,
                                     qbf, kbf, vbf, ksum, wmap, bmap);
  const int shbytes = (16*TROW + 16*5 + 16*17) * 4; // 263,808 B (< 320 KB WGP LDS)
  hipFuncSetAttribute((const void*)attn_kernel,
                      hipFuncAttributeMaxDynamicSharedMemorySize, shbytes);
  attn_kernel<<<1024, 256, shbytes, stream>>>(qbf, kbf, vbf, ksum, wmap, bmap, yv);
  conv_kernel<<<256, 256, 0, stream>>>(yv, x, w3b, linb, out);
}
